// ImprovedSelfAttention_25829933318605
// MI455X (gfx1250) — matrix operand order repaired and checked by reading, not compiled, not run on hardware
//
#include <hip/hip_runtime.h>
#include <cstdint>

typedef __bf16 bf16;
typedef __attribute__((ext_vector_type(16))) __bf16        v16bf;
typedef __attribute__((ext_vector_type(8)))  float         v8f;
typedef __attribute__((ext_vector_type(4)))  float         v4f;
typedef __attribute__((ext_vector_type(4)))  unsigned int  v4u;
typedef __attribute__((ext_vector_type(4)))  int           v4i;

union FragB16 { v4u u[2]; v16bf v; };

constexpr int BATCH = 4;
constexpr int CH    = 256;
constexpr int NHEAD = 8;
constexpr int DEPTH = 32;
constexpr int NTOK  = 2304;            // 48*48
constexpr int MROWS = BATCH * NTOK;    // 9216

// workspace layout (bytes)
constexpr size_t WT_BYTES  = (size_t)3 * CH * CH * 2;                  // 393216
constexpr size_t QKV_BYTES = (size_t)BATCH * NHEAD * NTOK * DEPTH * 2; // 4718592

#if defined(__has_builtin)
#if __has_builtin(__builtin_amdgcn_global_load_async_to_lds_b128) && \
    __has_builtin(__builtin_amdgcn_s_wait_asynccnt)
#define HAVE_ASYNC_LDS 1
#endif
#endif

// ---------------------------------------------------------------------------
// Kernel 0: convert W [c][d] f32 -> Wt [d][c] bf16 so WMMA B-frags load
// two runs of 8 contiguous K values per lane.
// ---------------------------------------------------------------------------
__global__ void prep_weights(const float* __restrict__ wq,
                             const float* __restrict__ wk,
                             const float* __restrict__ wv,
                             bf16* __restrict__ wT) {
  int idx = blockIdx.x * 256 + threadIdx.x;     // 0 .. 3*65536-1
  int mat = idx >> 16;
  int rem = idx & 65535;
  int d = rem >> 8;
  int c = rem & 255;
  const float* w = (mat == 0) ? wq : ((mat == 1) ? wk : wv);
  wT[idx] = (bf16)w[c * 256 + d];
}

// ---------------------------------------------------------------------------
// Kernel 1: fused QKV projection. One wave per 16x16 output tile.
// Output layout: [b, head, tok, depth] bf16 (Q pre-scaled by 1/sqrt(32)).
// ---------------------------------------------------------------------------
__global__ void qkv_proj(const float* __restrict__ x,
                         const bf16*  __restrict__ wT,
                         const float* __restrict__ bq,
                         const float* __restrict__ bk,
                         const float* __restrict__ bv,
                         bf16* __restrict__ qout,
                         bf16* __restrict__ kout,
                         bf16* __restrict__ vout) {
  const int lane = threadIdx.x & 31;
  const int rt = blockIdx.x;   // 0..575 (row tile over 9216 rows)
  const int ct = blockIdx.y;   // 0..15  (col tile over 256 cols)
  const int mz = blockIdx.z;   // 0=q 1=k 2=v

  const bf16*  w    = wT + (size_t)mz * CH * CH;
  const float* bias = (mz == 0) ? bq : ((mz == 1) ? bk : bv);
  bf16*        dst  = (mz == 0) ? qout : ((mz == 1) ? kout : vout);
  const float  scale = (mz == 0) ? 0.17677669529663687f : 1.0f;  // 1/sqrt(32)

  const int n   = lane & 15;
  const int hi  = lane >> 4;
  const int kbA = hi * 8;      // A-frag K base (per ISA 16-bit A layout)
  const int kbB = hi * 8;      // B-frag K base
  const int row = rt * 16 + n; // A row (lane = row)
  const int col = ct * 16 + n; // B col (lane = col)

  v8f acc = {};
#pragma unroll
  for (int k0 = 0; k0 < CH; k0 += 32) {
    // A fragment: x row, K = kbA+0..7 and kbA+16..23, f32 -> bf16
    const float* xr = x + (size_t)row * CH + k0 + kbA;
    v4f a0 = *(const v4f*)(xr);
    v4f a1 = *(const v4f*)(xr + 4);
    v4f a2 = *(const v4f*)(xr + 16);
    v4f a3 = *(const v4f*)(xr + 20);
    v16bf av;
#pragma unroll
    for (int j = 0; j < 4; ++j) {
      av[j]      = (bf16)a0[j];
      av[4 + j]  = (bf16)a1[j];
      av[8 + j]  = (bf16)a2[j];
      av[12 + j] = (bf16)a3[j];
    }
    // B fragment: Wt[col], K = kbB+0..7 and kbB+16..23
    FragB16 fb;
    const bf16* wr = w + (size_t)col * CH + k0 + kbB;
    fb.u[0] = *(const v4u*)(wr);
    fb.u[1] = *(const v4u*)(wr + 16);
    acc = __builtin_amdgcn_wmma_f32_16x16x32_bf16(false, av, false, fb.v,
                                                  (short)0, acc, false, false);
  }

  const float bb    = bias[col];
  const int   headv = col >> 5;
  const int   dep   = col & 31;
#pragma unroll
  for (int i = 0; i < 8; ++i) {
    int r    = rt * 16 + hi * 8 + i;   // C/D layout: VGPR i = row hi*8+i
    int bidx = r / NTOK;
    int tok  = r - bidx * NTOK;
    float val = (acc[i] + bb) * scale;
    dst[(((size_t)(bidx * NHEAD + headv)) * NTOK + tok) * DEPTH + dep] = (bf16)val;
  }
}

// ---------------------------------------------------------------------------
// Kernel 2: flash attention. 8 waves/WG, each wave owns a 16-row Q tile.
// K/V tiles of 32 tokens staged in LDS; online softmax; O += P*V via WMMA.
// ---------------------------------------------------------------------------
__global__ void __launch_bounds__(256) flash_attn(
    const bf16* __restrict__ qg, const bf16* __restrict__ kg,
    const bf16* __restrict__ vg, const float* __restrict__ x,
    const float* __restrict__ gamma_p, float* __restrict__ out) {
  __shared__ bf16 sK[32 * 32];        // [tok][depth]
  __shared__ bf16 sVT[32 * 32];       // [depth][tok]
  __shared__ bf16 sP[8][16 * 32];     // per-wave P staging [row][kvcol]

  const int tid  = threadIdx.x;
  const int lane = tid & 31;
  const int wid  = tid >> 5;
  const int bh   = blockIdx.y;                       // b*8 + head
  const int qbase = (blockIdx.x * 8 + wid) * 16;

  const int n  = lane & 15;
  const int hi = lane >> 4;

  // Q A-fragment: row = qbase+n, K(depth) = hi*16+0..15
  FragB16 qa;
  {
    const bf16* qr = qg + (((size_t)bh * NTOK) + qbase + n) * DEPTH + hi * 16;
    qa.u[0] = *(const v4u*)(qr);
    qa.u[1] = *(const v4u*)(qr + 8);
  }

  v8f O0 = {}, O1 = {};
  float mrow[8], lrow[8];
#pragma unroll
  for (int i = 0; i < 8; ++i) { mrow[i] = -1e30f; lrow[i] = 0.0f; }

  const bf16* kbase_g = kg + (size_t)bh * NTOK * DEPTH;
  const bf16* vbase_g = vg + (size_t)bh * NTOK * DEPTH;

  for (int kt = 0; kt < NTOK / 32; ++kt) {
    __syncthreads();  // previous iteration done reading sK/sVT

    if (tid < 128) {
      // K tile: 32 tok x 32 depth bf16 = 2048B contiguous -> LDS
      const bf16* src = kbase_g + (size_t)kt * 32 * DEPTH + tid * 8;
#if defined(HAVE_ASYNC_LDS)
      __builtin_amdgcn_global_load_async_to_lds_b128(
          (__attribute__((address_space(1))) v4i*)(uintptr_t)src,
          (__attribute__((address_space(3))) v4i*)(unsigned int)(uintptr_t)(&sK[tid * 8]),
          0, 0);
#else
      *(v4u*)(&sK[tid * 8]) = *(const v4u*)src;
#endif
    } else {
      // V tile transposed into LDS: sVT[d][tok]
      int t2    = tid - 128;       // 0..127
      int tok   = t2 >> 2;         // 0..31
      int dbase = (t2 & 3) * 8;
      const bf16* src = vbase_g + ((size_t)kt * 32 + tok) * DEPTH + dbase;
      v4u raw = *(const v4u*)src;
      const bf16* e = (const bf16*)&raw;
#pragma unroll
      for (int j = 0; j < 8; ++j) sVT[(dbase + j) * 32 + tok] = e[j];
    }
    if (kt + 1 < NTOK / 32 && tid < 32)
      __builtin_prefetch(kbase_g + (size_t)(kt + 1) * 32 * DEPTH + tid * 32, 0, 0);
#if defined(HAVE_ASYNC_LDS)
    __builtin_amdgcn_s_wait_asynccnt(0);
#endif
    __syncthreads();

    // Kt B-frags: lane=kv col, K(depth)=hi*16..+15 contiguous
    FragB16 bk0, bk1;
    {
      const bf16* p0 = &sK[n * DEPTH + hi * 16];
      bk0.u[0] = *(const v4u*)p0; bk0.u[1] = *(const v4u*)(p0 + 8);
      const bf16* p1 = &sK[(16 + n) * DEPTH + hi * 16];
      bk1.u[0] = *(const v4u*)p1; bk1.u[1] = *(const v4u*)(p1 + 8);
    }
    v8f zero = {};
    v8f S0 = __builtin_amdgcn_wmma_f32_16x16x32_bf16(false, qa.v, false, bk0.v,
                                                     (short)0, zero, false, false);
    v8f S1 = __builtin_amdgcn_wmma_f32_16x16x32_bf16(false, qa.v, false, bk1.v,
                                                     (short)0, zero, false, false);

    // online softmax: rows live as VGPR index, cols across 16 lanes
    float corr[8];
#pragma unroll
    for (int i = 0; i < 8; ++i) {
      float xm = S0[i] > S1[i] ? S0[i] : S1[i];
#pragma unroll
      for (int off = 1; off < 16; off <<= 1) {
        float o = __shfl_xor(xm, off);
        xm = xm > o ? xm : o;
      }
      float mnew = mrow[i] > xm ? mrow[i] : xm;
      corr[i] = __expf(mrow[i] - mnew);
      float p0 = __expf(S0[i] - mnew);
      float p1 = __expf(S1[i] - mnew);
      S0[i] = p0; S1[i] = p1;
      float rs = p0 + p1;
#pragma unroll
      for (int off = 1; off < 16; off <<= 1) rs += __shfl_xor(rs, off);
      lrow[i] = lrow[i] * corr[i] + rs;
      mrow[i] = mnew;
      O0[i] *= corr[i];
      O1[i] *= corr[i];
    }

    // C-layout -> A-layout for P via per-wave LDS (LDS ops in-order per wave)
    bf16* pw = &sP[wid][0];
#pragma unroll
    for (int i = 0; i < 8; ++i) {
      int r = hi * 8 + i;
      pw[r * 32 + n]      = (bf16)S0[i];
      pw[r * 32 + 16 + n] = (bf16)S1[i];
    }

    // V B-frags: lane=depth col, K(tok)=hi*16..+15 contiguous in sVT
    FragB16 bv0, bv1;
    {
      const bf16* p0 = &sVT[n * 32 + hi * 16];
      bv0.u[0] = *(const v4u*)p0; bv0.u[1] = *(const v4u*)(p0 + 8);
      const bf16* p1 = &sVT[(16 + n) * 32 + hi * 16];
      bv1.u[0] = *(const v4u*)p1; bv1.u[1] = *(const v4u*)(p1 + 8);
    }

    FragB16 pa;
    {
      const bf16* pr = &pw[n * 32 + hi * 16];
      pa.u[0] = *(const v4u*)(pr);
      pa.u[1] = *(const v4u*)(pr + 8);
    }

    O0 = __builtin_amdgcn_wmma_f32_16x16x32_bf16(false, pa.v, false, bv0.v,
                                                 (short)0, O0, false, false);
    O1 = __builtin_amdgcn_wmma_f32_16x16x32_bf16(false, pa.v, false, bv1.v,
                                                 (short)0, O1, false, false);
  }

  // epilogue: out = gamma * (O / l) + x
  const float g = *gamma_p;
  const int b    = bh >> 3;
  const int head = bh & 7;
#pragma unroll
  for (int i = 0; i < 8; ++i) {
    int tok = qbase + hi * 8 + i;
    size_t r = (size_t)b * NTOK + tok;
    float inv = 1.0f / lrow[i];
    int c0 = head * DEPTH + n;
    int c1 = c0 + 16;
    out[r * CH + c0] = g * (O0[i] * inv) + x[r * CH + c0];
    out[r * CH + c1] = g * (O1[i] * inv) + x[r * CH + c1];
  }
}

// ---------------------------------------------------------------------------
extern "C" void kernel_launch(void* const* d_in, const int* in_sizes, int n_in,
                              void* d_out, int out_size, void* d_ws, size_t ws_size,
                              hipStream_t stream) {
  (void)in_sizes; (void)n_in; (void)out_size; (void)ws_size;
  const float* x  = (const float*)d_in[0];
  const float* wq = (const float*)d_in[1];
  const float* bq = (const float*)d_in[2];
  const float* wk = (const float*)d_in[3];
  const float* bk = (const float*)d_in[4];
  const float* wv = (const float*)d_in[5];
  const float* bv = (const float*)d_in[6];
  const float* gm = (const float*)d_in[7];
  float* out = (float*)d_out;

  char* ws = (char*)d_ws;
  bf16* wT = (bf16*)(ws);
  bf16* q  = (bf16*)(ws + WT_BYTES);
  bf16* k  = (bf16*)(ws + WT_BYTES + QKV_BYTES);
  bf16* v  = (bf16*)(ws + WT_BYTES + 2 * QKV_BYTES);

  prep_weights<<<dim3(768), dim3(256), 0, stream>>>(wq, wk, wv, wT);
  qkv_proj<<<dim3(MROWS / 16, CH / 16, 3), dim3(32), 0, stream>>>(
      x, wT, bq, bk, bv, q, k, v);
  flash_attn<<<dim3(NTOK / 128, BATCH * NHEAD), dim3(256), 0, stream>>>(
      q, k, v, x, gm, out);
}
